// encoder_41128606826563
// MI455X (gfx1250) — compile-verified, matrix operand order checked by reading, not run on hardware
//
#include <hip/hip_runtime.h>
#include <hip/hip_bf16.h>

typedef __attribute__((ext_vector_type(16))) _Float16 v16h;
typedef __attribute__((ext_vector_type(8)))  float    v8f;

#define F 128          // feature width (both in and hidden)
#define BN_EPS 1e-5f

// ---------------------------------------------------------------------------
// K-index mapping for the 16-bit WMMA A/B operand layout (ISA 7.12.2):
// lane 0-15 -> M(or N)=lane, lane 16-31 -> same M/N but K+8 (khalf).
// half h (0..15) within the 32B/lane: v = h/2, j = h%2,
// K = (v/4)*16 + (v%4)*2 + khalf*8 + j   (within a K=32 slab)
// ---------------------------------------------------------------------------
__device__ __forceinline__ int wmma_k_of(int h, int lane, int ktile) {
    int v = h >> 1, j = h & 1, khalf = lane >> 4;
    return ktile * 32 + (v >> 2) * 16 + (v & 3) * 2 + khalf * 8 + j;
}

// ---- misc init: deg=1 (self loop weight), BN stats = 0 ---------------------
__global__ void k_init_misc(float* deg, float* stats, int n_nodes) {
    int i = blockIdx.x * blockDim.x + threadIdx.x;
    if (i < n_nodes) deg[i] = 1.0f;
    if (i < 2 * F)   stats[i] = 0.0f;
}

// ---- degree accumulation: deg[dst] += w[e] --------------------------------
__global__ void k_deg(const int* ei, const float* w, float* deg, int E) {
    int e = blockIdx.x * blockDim.x + threadIdx.x;
    if (e >= E) return;
    atomicAdd(deg + ei[E + e], w[e]);
}

// ---- dinv[i] = deg>0 ? rsqrt(deg) : 0 -------------------------------------
__global__ void k_dinv(const float* deg, float* dinv, int n_nodes) {
    int i = blockIdx.x * blockDim.x + threadIdx.x;
    if (i >= n_nodes) return;
    float d = deg[i];
    dinv[i] = (d > 0.0f) ? rsqrtf(d) : 0.0f;
}

// ---- norm[e] = dinv[src]*w*dinv[dst] --------------------------------------
__global__ void k_norm(const int* ei, const float* w, const float* dinv,
                       float* norm, int E) {
    int e = blockIdx.x * blockDim.x + threadIdx.x;
    if (e >= E) return;
    norm[e] = dinv[ei[e]] * w[e] * dinv[ei[E + e]];
}

// ---- BN stats: per-column sum / sumsq over a 256-row slab ------------------
__global__ void k_bn_stats(const float* x, float* stats, int n_nodes) {
    int f  = threadIdx.x;               // 0..127
    int r0 = blockIdx.x * 256;
    int r1 = r0 + 256 < n_nodes ? r0 + 256 : n_nodes;
    float s = 0.f, s2 = 0.f;
    for (int r = r0; r < r1; ++r) {
        float v = x[(size_t)r * F + f];
        s += v; s2 += v * v;
    }
    atomicAdd(&stats[f], s);
    atomicAdd(&stats[F + f], s2);
}

// ---- BN normalize -> x_nor -------------------------------------------------
__global__ void k_bn_norm(const float* x, const float* stats,
                          const float* gamma, const float* beta,
                          float* out, size_t n_elems, float invN) {
    size_t idx = (size_t)blockIdx.x * blockDim.x + threadIdx.x;
    if (idx >= n_elems) return;
    int f = (int)(idx & (F - 1));
    float mean = stats[f] * invN;
    float var  = stats[F + f] * invN - mean * mean;
    out[idx] = (x[idx] - mean) * rsqrtf(var + BN_EPS) * gamma[f] + beta[f];
}

// ---- pack fp32 weights [K=128][N=128] into WMMA B-operand layout -----------
// Wh layout per layer: [ntile(8)][ktile(4)][lane(32)][16 halves]
__global__ void k_pack_w(const float* W1, const float* W2, const float* W3,
                         _Float16* Wh) {
    int t = blockIdx.x * blockDim.x + threadIdx.x;
    if (t >= 3 * 16384) return;
    int L = t / 16384, r = t % 16384;
    int h = r & 15, lane = (r >> 4) & 31, kt = (r >> 9) & 3, ntile = r >> 11;
    int n = ntile * 16 + (lane & 15);
    int k = wmma_k_of(h, lane, kt);
    const float* W = (L == 0) ? W1 : (L == 1) ? W2 : W3;
    Wh[t] = (_Float16)W[k * F + n];
}

// ---- pack fp32 activations [N][128] into WMMA A-operand layout (+ReLU) -----
// A layout: [rowtile][ktile(4)][lane(32)][16 halves]
__global__ void k_pack_a(const float* in, _Float16* out, size_t n_half, int relu) {
    size_t t = (size_t)blockIdx.x * blockDim.x + threadIdx.x;
    if (t >= n_half) return;
    int h = (int)(t & 15), lane = (int)((t >> 4) & 31), kt = (int)((t >> 9) & 3);
    size_t rt = t >> 11;
    int m = lane & 15;
    int k = wmma_k_of(h, lane, kt);
    float val = in[(rt * 16 + m) * F + k];
    if (relu) val = fmaxf(val, 0.0f);
    out[t] = (_Float16)val;
}

// ---- fused WMMA GEMM + self-loop/bias epilogue -----------------------------
// t[N x 128]   = A(f16 packed) x W(f16 packed), f32 accumulate   (gather src)
// out[N x 128] = bias + dinv[row]^2 * t                          (agg init)
// 1 block = 16 rows; wave w (0..7) owns 16-column tile w; 4 WMMAs along K=128
__global__ void k_gemm(const v16h* __restrict__ A, const v16h* __restrict__ Wh,
                       float* __restrict__ t, float* __restrict__ out,
                       const float* __restrict__ dinv,
                       const float* __restrict__ bias) {
    int rt   = blockIdx.x;
    int wave = threadIdx.x >> 5;     // ntile
    int lane = threadIdx.x & 31;
    v8f acc = {};
#pragma unroll
    for (int kt = 0; kt < 4; ++kt) {
        v16h a = A[((size_t)rt * 4 + kt) * 32 + lane];
        v16h b = Wh[((size_t)wave * 4 + kt) * 32 + lane];
        acc = __builtin_amdgcn_wmma_f32_16x16x32_f16(
            /*neg_a=*/false, a, /*neg_b=*/false, b,
            /*c_mod=*/(short)0, acc, /*reuse_a=*/false, /*reuse_b=*/false);
    }
    // C layout: VGPR i -> row = i + 8*(lane/16), col = lane%16
    int m0  = (lane >> 4) * 8;
    int col = wave * 16 + (lane & 15);
    size_t row0 = (size_t)rt * 16;
    float b_ = bias[col];
#pragma unroll
    for (int i = 0; i < 8; ++i) {
        size_t row = row0 + m0 + i;
        float  v   = acc[i];
        float  di  = dinv[row];
        t[row * F + col]   = v;
        out[row * F + col] = b_ + di * di * v;
    }
}

// ---- edge scatter: out[dst] += norm[e] * t[src]; 1 wave per edge -----------
__global__ void k_edge_scatter(const int* __restrict__ ei,
                               const float* __restrict__ norm,
                               const float* __restrict__ t,
                               float* __restrict__ out, int E) {
    int gid  = blockIdx.x * blockDim.x + threadIdx.x;
    int e    = gid >> 5;
    int lane = gid & 31;
    if (e >= E) return;
    int   s = ei[e];
    int   d = ei[E + e];
    float w = norm[e];
    float4 v = reinterpret_cast<const float4*>(t + (size_t)s * F)[lane];
    float* o = out + (size_t)d * F + lane * 4;
    atomicAdd(o + 0, w * v.x);
    atomicAdd(o + 1, w * v.y);
    atomicAdd(o + 2, w * v.z);
    atomicAdd(o + 3, w * v.w);
}

// ---------------------------------------------------------------------------
extern "C" void kernel_launch(void* const* d_in, const int* in_sizes, int n_in,
                              void* d_out, int out_size, void* d_ws, size_t ws_size,
                              hipStream_t stream) {
    const float* x     = (const float*)d_in[0];
    const int*   ei    = (const int*)d_in[1];
    const float* ew    = (const float*)d_in[3];
    const float* gamma = (const float*)d_in[4];
    const float* beta  = (const float*)d_in[5];
    const float* W1    = (const float*)d_in[6];
    const float* b1    = (const float*)d_in[7];
    const float* W2    = (const float*)d_in[8];
    const float* b2    = (const float*)d_in[9];
    const float* W3    = (const float*)d_in[10];
    const float* b3    = (const float*)d_in[11];

    const int    N  = in_sizes[0] / F;     // 100000
    const int    E  = in_sizes[3];         // 1600000
    const size_t NF = (size_t)N * F;       // 12.8M elements

    float* out_h   = (float*)d_out;        // layer-3 result
    float* out_nor = (float*)d_out + NF;   // x_nor

    // ---- workspace carving (≈135 MB, 256B aligned) ----
    char* ws = (char*)d_ws;
    size_t off = 0;
    auto carve = [&](size_t bytes) { void* p = ws + off; off = (off + bytes + 255) & ~(size_t)255; return p; };
    _Float16* A    = (_Float16*)carve(NF * sizeof(_Float16));    // 25.6 MB packed activations
    float*    t    = (float*)   carve(NF * sizeof(float));       // 51.2 MB GEMM out
    float*    hbuf = (float*)   carve(NF * sizeof(float));       // 51.2 MB inter-layer
    float*    nrm  = (float*)   carve((size_t)E * sizeof(float));// 6.4 MB edge norms
    float*    deg  = (float*)   carve((size_t)N * sizeof(float));
    float*    dinv = (float*)   carve((size_t)N * sizeof(float));
    float*    stats= (float*)   carve(2 * F * sizeof(float));
    _Float16* Wh   = (_Float16*)carve(3 * 16384 * sizeof(_Float16));
    (void)ws_size; (void)n_in; (void)out_size;

    const int BLK = 256;
    const unsigned gN    = (N + BLK - 1) / BLK;
    const unsigned gE    = (E + BLK - 1) / BLK;
    const unsigned gNF   = (unsigned)((NF + BLK - 1) / BLK);
    const unsigned gEdge = (unsigned)(((size_t)E * 32 + BLK - 1) / BLK);
    const unsigned gGemm = (unsigned)(N / 16);                  // 6250 row tiles
    const unsigned gBN   = (N + 255) / 256;

    // --- graph norm + BN stats (independent of layers) ---
    k_init_misc<<<gN, BLK, 0, stream>>>(deg, stats, N);
    k_deg      <<<gE, BLK, 0, stream>>>(ei, ew, deg, E);
    k_dinv     <<<gN, BLK, 0, stream>>>(deg, dinv, N);
    k_norm     <<<gE, BLK, 0, stream>>>(ei, ew, dinv, nrm, E);
    k_bn_stats <<<gBN, F, 0, stream>>>(x, stats, N);
    k_bn_norm  <<<gNF, BLK, 0, stream>>>(x, stats, gamma, beta, out_nor, NF, 1.0f / (float)N);

    // --- weights to WMMA B layout (f16) ---
    k_pack_w<<<(3 * 16384 + BLK - 1) / BLK, BLK, 0, stream>>>(W1, W2, W3, Wh);

    // --- layer 1: x -> hbuf ---
    k_pack_a      <<<gNF, BLK, 0, stream>>>(x, A, NF, 0);
    k_gemm        <<<gGemm, BLK, 0, stream>>>((const v16h*)A, (const v16h*)(Wh + 0 * 16384), t, hbuf, dinv, b1);
    k_edge_scatter<<<gEdge, BLK, 0, stream>>>(ei, nrm, t, hbuf, E);

    // --- layer 2: relu(hbuf) -> hbuf ---
    k_pack_a      <<<gNF, BLK, 0, stream>>>(hbuf, A, NF, 1);
    k_gemm        <<<gGemm, BLK, 0, stream>>>((const v16h*)A, (const v16h*)(Wh + 1 * 16384), t, hbuf, dinv, b2);
    k_edge_scatter<<<gEdge, BLK, 0, stream>>>(ei, nrm, t, hbuf, E);

    // --- layer 3: relu(hbuf) -> d_out (no relu on output) ---
    k_pack_a      <<<gNF, BLK, 0, stream>>>(hbuf, A, NF, 1);
    k_gemm        <<<gGemm, BLK, 0, stream>>>((const v16h*)A, (const v16h*)(Wh + 2 * 16384), t, out_h, dinv, b3);
    k_edge_scatter<<<gEdge, BLK, 0, stream>>>(ei, nrm, t, out_h, E);
}
